// GNNExplainerAgg_44770739093934
// MI455X (gfx1250) — matrix operand order repaired and checked by reading, not compiled, not run on hardware
//
#include <hip/hip_runtime.h>
#include <math.h>

#define N_NODES 50000
#define N_EDGES 800000
#define D 128
#define NT_NODE (N_NODES / 16)   // 3125 node tiles

typedef __attribute__((ext_vector_type(16))) __bf16        v16bf;
typedef __attribute__((ext_vector_type(8)))  float         v8f;
typedef __attribute__((ext_vector_type(4)))  unsigned int  u32x4;

union BfTile { u32x4 u[2]; v16bf v; unsigned short s[16]; };

__device__ __forceinline__ unsigned short f2bf(float f) {
  unsigned int u = __float_as_uint(f);
  u += 0x7FFFu + ((u >> 16) & 1u);   // round-to-nearest-even
  return (unsigned short)(u >> 16);
}

// ---------------- prep kernels ----------------

__global__ void k_cvt_x(const float* __restrict__ x, unsigned short* __restrict__ xbf, int n) {
  int i = blockIdx.x * 256 + threadIdx.x;
  if (i < n) xbf[i] = f2bf(x[i]);
}

// Pack W1 [256,128] f32 into bf16 B-register layout.
// tile = (half*8 + nt)*4 + kb ; within tile: lane (0..31) * 16 + j (0..15)
// lane<16 : n = nt*16+lane,      k = kb*32 + (j&7) + (j>=8 ? 16 : 0)
// lane>=16: n = nt*16+(lane-16), k = kb*32 + 8 + (j&7) + (j>=8 ? 16 : 0)
__global__ void k_pack_w1(const float* __restrict__ W1, unsigned short* __restrict__ w1p) {
  int i = blockIdx.x * 256 + threadIdx.x;            // 32768 threads
  int tile = i >> 9, lane = (i >> 4) & 31, j = i & 15;
  int half = tile >> 5, nt = (tile >> 2) & 7, kb = tile & 3;
  int k = kb * 32 + ((lane >= 16) ? 8 : 0) + (j & 7) + ((j >= 8) ? 16 : 0);
  int n = nt * 16 + (lane & 15);
  w1p[i] = f2bf(W1[(size_t)(half * 128 + k) * 128 + n]);
}

__global__ void k_pack_wl(const float* __restrict__ Wl, unsigned short* __restrict__ wlp) {
  int i = blockIdx.x * 256 + threadIdx.x;            // 16384 threads
  int tile = i >> 9, lane = (i >> 4) & 31, j = i & 15;
  int nt = tile >> 2, kb = tile & 3;
  int k = kb * 32 + ((lane >= 16) ? 8 : 0) + (j & 7) + ((j >= 8) ? 16 : 0);
  int n = nt * 16 + (lane & 15);
  wlp[i] = f2bf(Wl[(size_t)k * 128 + n]);
}

__global__ void k_zero(float* __restrict__ segsum, float* __restrict__ agg) {
  int i = blockIdx.x * 256 + threadIdx.x;            // covers N_NODES*D exactly
  if (i < N_NODES) segsum[i] = 0.0f;
  agg[i] = 0.0f;
}

// ---------------- edge MLP + mask (the 52 GFLOP hot loop) ----------------

__device__ __forceinline__ void edge_epilogue(
    const v8f C[8], int ebase, int m, int hi,
    const float* __restrict__ b1, const float* __restrict__ W2, float b2v,
    float* __restrict__ maskg, float* __restrict__ segsum, const int* __restrict__ row_g) {
  float partial[8] = {0.f, 0.f, 0.f, 0.f, 0.f, 0.f, 0.f, 0.f};
#pragma unroll
  for (int nt = 0; nt < 8; nt++) {
    int n = nt * 16 + m;
    float bias = b1[n];
    float w2v = W2[n];
#pragma unroll
    for (int r = 0; r < 8; r++) {
      float h = fmaxf(C[nt][r] + bias, 0.f);
      partial[r] += h * w2v;
    }
  }
#pragma unroll
  for (int r = 0; r < 8; r++) {
    float p = partial[r];
    p += __shfl_xor(p, 1, 32);
    p += __shfl_xor(p, 2, 32);
    p += __shfl_xor(p, 4, 32);
    p += __shfl_xor(p, 8, 32);                       // full sum within each 16-lane half
    float mv = 1.f / (1.f + __expf(-(p + b2v)));     // sigmoid
    if (m == r) {                                    // lanes r and r+16 own edges r, r+8
      int e = ebase + r + (hi << 3);
      maskg[e] = mv;
      atomicAdd(&segsum[row_g[e]], mv);
    }
  }
}

// One wave = 32 edges (two 16-row M-tiles): each B tile from LDS feeds 2 WMMAs.
__global__ __launch_bounds__(256) void k_edge(
    const unsigned short* __restrict__ xbf, const unsigned short* __restrict__ w1p,
    const int* __restrict__ row_g, const int* __restrict__ col_g,
    const float* __restrict__ b1, const float* __restrict__ W2, const float* __restrict__ b2,
    float* __restrict__ maskg, float* __restrict__ segsum) {
  __shared__ u32x4 sW1[4096];                        // 64 KB packed W1
  int t = threadIdx.x;
  const u32x4* wp = (const u32x4*)w1p;
#pragma unroll
  for (int i = 0; i < 16; i++) sW1[t + i * 256] = wp[t + i * 256];
  __syncthreads();

  int pair = blockIdx.x * 8 + (t >> 5);              // one wave per 32-edge pair
  int lane = t & 31;
  int m = lane & 15;
  int hi = lane >> 4;
  int e0 = pair * 32;

  int r0 = row_g[e0 + m],      cc0 = col_g[e0 + m];
  int r1 = row_g[e0 + 16 + m], cc1 = col_g[e0 + 16 + m];

  v8f C0[8], C1[8];
#pragma unroll
  for (int nt = 0; nt < 8; nt++) {
    C0[nt] = (v8f){0.f, 0.f, 0.f, 0.f, 0.f, 0.f, 0.f, 0.f};
    C1[nt] = (v8f){0.f, 0.f, 0.f, 0.f, 0.f, 0.f, 0.f, 0.f};
  }

#pragma unroll
  for (int kb = 0; kb < 4; kb++) {
    int koff = kb * 32 + (hi ? 8 : 0);
    BfTile Ac0, An0, Ac1, An1;                        // A tiles for this K-chunk
    {
      const u32x4* p0 = (const u32x4*)(xbf + (size_t)r0  * D + koff);
      Ac0.u[0] = p0[0]; Ac0.u[1] = p0[2];
      const u32x4* p1 = (const u32x4*)(xbf + (size_t)cc0 * D + koff);
      An0.u[0] = p1[0]; An0.u[1] = p1[2];
      const u32x4* p2 = (const u32x4*)(xbf + (size_t)r1  * D + koff);
      Ac1.u[0] = p2[0]; Ac1.u[1] = p2[2];
      const u32x4* p3 = (const u32x4*)(xbf + (size_t)cc1 * D + koff);
      An1.u[0] = p3[0]; An1.u[1] = p3[2];
    }
#pragma unroll
    for (int g = 0; g < 2; g++) {                     // nt groups of 4
      BfTile B[4];
#pragma unroll
      for (int q = 0; q < 4; q++) {                   // ds clause: half-0 B burst
        const u32x4* pb = &sW1[(size_t)((0 * 8 + g * 4 + q) * 4 + kb) * 64 + lane * 2];
        B[q].u[0] = pb[0]; B[q].u[1] = pb[1];
      }
#pragma unroll
      for (int q = 0; q < 4; q++) {                   // WMMA burst (2 per B tile)
        C0[g * 4 + q] = __builtin_amdgcn_wmma_f32_16x16x32_bf16(false, Ac0.v, false, B[q].v, (short)0, C0[g * 4 + q], false, false);
        C1[g * 4 + q] = __builtin_amdgcn_wmma_f32_16x16x32_bf16(false, Ac1.v, false, B[q].v, (short)0, C1[g * 4 + q], false, false);
      }
#pragma unroll
      for (int q = 0; q < 4; q++) {                   // ds clause: half-1 B burst
        const u32x4* pb = &sW1[(size_t)((1 * 8 + g * 4 + q) * 4 + kb) * 64 + lane * 2];
        B[q].u[0] = pb[0]; B[q].u[1] = pb[1];
      }
#pragma unroll
      for (int q = 0; q < 4; q++) {
        C0[g * 4 + q] = __builtin_amdgcn_wmma_f32_16x16x32_bf16(false, An0.v, false, B[q].v, (short)0, C0[g * 4 + q], false, false);
        C1[g * 4 + q] = __builtin_amdgcn_wmma_f32_16x16x32_bf16(false, An1.v, false, B[q].v, (short)0, C1[g * 4 + q], false, false);
      }
    }
  }

  float b2v = b2[0];
  edge_epilogue(C0, e0,      m, hi, b1, W2, b2v, maskg, segsum, row_g);
  edge_epilogue(C1, e0 + 16, m, hi, b1, W2, b2v, maskg, segsum, row_g);
}

// ---------------- normalized neighbor aggregation ----------------

__global__ __launch_bounds__(128) void k_agg(
    const float* __restrict__ x, const int* __restrict__ row_g, const int* __restrict__ col_g,
    const float* __restrict__ maskg, const float* __restrict__ segsum, float* __restrict__ agg) {
  int e = blockIdx.x;
  int f = threadIdx.x;
  int r = row_g[e], c = col_g[e];
  float w = maskg[e] / (segsum[r] + 1e-9f);
  atomicAdd(&agg[(size_t)r * D + f], w * x[(size_t)c * D + f]);
}

// ---------------- blend + final linear + relu ----------------

__global__ __launch_bounds__(256) void k_final(
    const float* __restrict__ x, const float* __restrict__ agg,
    const float* __restrict__ segsum, const unsigned short* __restrict__ wlp,
    const float* __restrict__ blg, float* __restrict__ out) {
  __shared__ u32x4 sWl[2048];                        // 32 KB packed Wl
  int t = threadIdx.x;
  const u32x4* wp = (const u32x4*)wlp;
#pragma unroll
  for (int i = 0; i < 8; i++) sWl[t + i * 256] = wp[t + i * 256];
  __syncthreads();

  int tile = blockIdx.x * 8 + (t >> 5);
  if (tile >= NT_NODE) return;                       // wave-uniform
  int lane = t & 31, m = lane & 15, hi = lane >> 4;
  int node = tile * 16 + m;
  const float* xr = x + (size_t)node * D;
  const float* ar = agg + (size_t)node * D;
  float s = segsum[node];                            // >0 iff deg>0 (sigmoid>0)
  float wx = (s > 0.f) ? 0.5f : 1.0f;
  float wa = (s > 0.f) ? 0.5f : 0.0f;

  BfTile A[4];
#pragma unroll
  for (int kb = 0; kb < 4; kb++) {
    int koff = kb * 32 + (hi ? 8 : 0);
#pragma unroll
    for (int j = 0; j < 8; j++) {
      float e0 = wx * xr[koff + j]      + wa * ar[koff + j];
      float e1 = wx * xr[koff + 16 + j] + wa * ar[koff + 16 + j];
      A[kb].s[j]     = f2bf(e0);
      A[kb].s[8 + j] = f2bf(e1);
    }
  }

  v8f C[8];
#pragma unroll
  for (int nt = 0; nt < 8; nt++) C[nt] = (v8f){0.f, 0.f, 0.f, 0.f, 0.f, 0.f, 0.f, 0.f};

#pragma unroll
  for (int kb = 0; kb < 4; kb++) {
    BfTile B[8];
#pragma unroll
    for (int nt = 0; nt < 8; nt++) {                 // ds clause: 8-tile B burst
      const u32x4* pb = &sWl[(size_t)(nt * 4 + kb) * 64 + lane * 2];
      B[nt].u[0] = pb[0]; B[nt].u[1] = pb[1];
    }
#pragma unroll
    for (int nt = 0; nt < 8; nt++) {                 // WMMA burst
      C[nt] = __builtin_amdgcn_wmma_f32_16x16x32_bf16(false, A[kb].v, false, B[nt].v, (short)0, C[nt], false, false);
    }
  }

#pragma unroll
  for (int nt = 0; nt < 8; nt++) {
    int n = nt * 16 + m;
    float bb = blg[n];
#pragma unroll
    for (int r = 0; r < 8; r++) {
      float v = fmaxf(C[nt][r] + bb, 0.f);
      int nd = tile * 16 + r + (hi << 3);
      out[(size_t)nd * D + n] = v;
    }
  }
}

// ---------------- launcher ----------------

extern "C" void kernel_launch(void* const* d_in, const int* in_sizes, int n_in,
                              void* d_out, int out_size, void* d_ws, size_t ws_size,
                              hipStream_t stream) {
  const float* x  = (const float*)d_in[0];
  const int*   ei = (const int*)d_in[1];
  const float* W1 = (const float*)d_in[2];
  const float* b1 = (const float*)d_in[3];
  const float* W2 = (const float*)d_in[4];
  const float* b2 = (const float*)d_in[5];
  const float* Wl = (const float*)d_in[6];
  const float* bl = (const float*)d_in[7];
  const int* row = ei;
  const int* col = ei + N_EDGES;

  char* ws = (char*)d_ws;
  unsigned short* xbf    = (unsigned short*)(ws + 0);         // 12,800,000 B
  unsigned short* w1p    = (unsigned short*)(ws + 12800000);  //     65,536 B
  unsigned short* wlp    = (unsigned short*)(ws + 12865536);  //     32,768 B
  float*          segsum = (float*)(ws + 12898304);           //    200,192 B (padded)
  float*          agg    = (float*)(ws + 13098496);           // 25,600,000 B
  float*          maskg  = (float*)(ws + 38698496);           //  3,200,000 B
  float*          out    = (float*)d_out;

  k_cvt_x  <<<25000, 256, 0, stream>>>(x, xbf, N_NODES * D);  // 6.4M elems exactly
  k_pack_w1<<<  128, 256, 0, stream>>>(W1, w1p);
  k_pack_wl<<<   64, 256, 0, stream>>>(Wl, wlp);
  k_zero   <<<25000, 256, 0, stream>>>(segsum, agg);
  k_edge   <<< 3125, 256, 0, stream>>>(xbf, w1p, row, col, b1, W2, b2, maskg, segsum);
  k_agg    <<<N_EDGES, 128, 0, stream>>>(x, row, col, maskg, segsum, agg);
  k_final  <<<  391, 256, 0, stream>>>(x, agg, segsum, wlp, bl, out);
}
